// LSTM_82892868812973
// MI455X (gfx1250) — matrix operand order repaired and checked by reading, not compiled, not run on hardware
//
#include <hip/hip_runtime.h>

typedef __attribute__((ext_vector_type(16))) _Float16 v16h;
typedef __attribute__((ext_vector_type(8)))  _Float16 v8h;
typedef __attribute__((ext_vector_type(8)))  float    v8f;

#define TPB     256
#define T_STEPS 28
#define IN_DIM  28
#define H_DIM   128
#define KREAL   (H_DIM + IN_DIM)   // 156; K=156 carries the bias (g column == 1.0)
#define LDSK    168                 // padded LDS row stride (halves); 336B, 16B aligned
#define MROWS   16

// Fast sigmoid/tanh: one v_exp_f32 + one v_rcp_f32 each (TRANS pipe, co-executes
// with WMMA). Avoids the IEEE v_div_scale/div_fmas expansion.
__device__ __forceinline__ float sigmoid_f(float x) {
    return __builtin_amdgcn_rcpf(1.0f + __expf(-x));
}

__device__ __forceinline__ float tanh_f(float x) {
    x = fminf(fmaxf(x, -15.0f), 15.0f);
    const float e = __expf(2.0f * x);
    return (e - 1.0f) * __builtin_amdgcn_rcpf(e + 1.0f);
}

__global__ __launch_bounds__(TPB) void lstm_wmma_kernel(
    const float* __restrict__ x,
    const float* __restrict__ Wf, const float* __restrict__ bf,
    const float* __restrict__ Wi, const float* __restrict__ bi,
    const float* __restrict__ Wc, const float* __restrict__ bc,
    const float* __restrict__ Wo, const float* __restrict__ bo,
    float* __restrict__ out)
{
    __shared__ _Float16 gbuf[MROWS * LDSK];   // [16 rows x 160(+pad) K] f16 "g" matrix

    const int tid  = threadIdx.x;
    const int wave = tid >> 5;
    const int lane = tid & 31;
    const int half = lane >> 4;      // half-wave select
    const int l16  = lane & 15;
    const int colbase = wave * 16;   // this wave's 16 output columns
    const long row0 = (long)blockIdx.x * MROWS;

    // ---------- preload weights (B-fragments, f16) into registers ----------
    // B frags, contiguous K order used here (32x16): lanes 0-15 hold N=lane, K=0..15; lanes 16-31 hold
    // N=lane-16, K=16..31; K packed 2-per-VGPR over 8 VGPRs (16 contiguous halves).
    // out = g @ W^T  =>  B[k][n] = W[n][k]  => lane reads 16 contiguous W-row floats.
    // Bias fold: B[156][n] = bias[n]; the g matrix carries 1.0 at K=156, so the
    // bias rides through the WMMA and accumulators can start from inline zero.
    const float* Wg[4]  = {Wf, Wi, Wc, Wo};
    const float* bgp[4] = {bf, bi, bc, bo};
    const int n = colbase + l16;     // output neuron index owned by this lane

    v16h Bw[4][5];
#pragma unroll
    for (int g = 0; g < 4; ++g) {
        const float* Wrow = Wg[g] + (long)n * KREAL;
        const float  bv   = bgp[g][n];
#pragma unroll
        for (int k = 0; k < 5; ++k) {
            const int kb = k * 32 + half * 16;
            v16h frag;
#pragma unroll
            for (int j = 0; j < 16; ++j) {
                const int kk = kb + j;
                float v;
                if (kk < KREAL)       v = Wrow[kk];
                else if (kk == KREAL) v = bv;      // bias column (K=156)
                else                  v = 0.0f;
                frag[j] = (_Float16)v;
            }
            Bw[g][k] = frag;
        }
    }

    // ---------- zero LDS (h region starts as h0 = 0, pad region stays 0) ----------
    for (int e = tid; e < MROWS * LDSK; e += TPB) gbuf[e] = (_Float16)0.0f;

    // cell state tile [16 rows x 16 cols] in C/D layout: lane l16 = col, vgpr r ->
    // row M = r + half*8
    v8f cstate;
#pragma unroll
    for (int r = 0; r < 8; ++r) cstate[r] = 0.0f;

    _Float16 hreg[8];   // h_{t-1} slice (f16), D layout
    float    hout[8];   // f32 copy of latest h (for final store)

    __syncthreads();

    for (int t = 0; t < T_STEPS; ++t) {
        // ---- store phase: x_t (28 values, then 1.0 bias column, then zeros) ----
        for (int e = tid; e < MROWS * 32; e += TPB) {
            const int m  = e >> 5;
            const int kk = e & 31;
            float v;
            if (kk < IN_DIM)       v = x[(row0 + m) * (long)(T_STEPS * IN_DIM)
                                         + t * IN_DIM + kk];
            else if (kk == IN_DIM) v = 1.0f;   // global K=156: bias multiplier
            else                   v = 0.0f;
            gbuf[m * LDSK + H_DIM + kk] = (_Float16)v;
        }
        if (t > 0) {
#pragma unroll
            for (int r = 0; r < 8; ++r) {
                const int m = r + half * 8;
                gbuf[m * LDSK + colbase + l16] = hreg[r];
            }
        }
        __syncthreads();

        // ---- load A fragments from LDS ----
        // A frags, contiguous K order used here (16x32): M = lane&15 for both halves; half 0 holds
        // K = 0..15, half 1 holds K = 16..31 (matches B fragments).
        v16h A[5];
        {
            const _Float16* base = &gbuf[(long)l16 * LDSK];
#pragma unroll
            for (int k = 0; k < 5; ++k) {
                const int kb = k * 32 + half * 16;
                v8h lo = *(const v8h*)(base + kb);        // 16B aligned
                v8h hi = *(const v8h*)(base + kb + 8);    // 16B aligned
                v16h a;
#pragma unroll
                for (int j = 0; j < 8; ++j) { a[j] = lo[j]; a[8 + j] = hi[j]; }
                A[k] = a;
            }
        }

        // ---- 4 gates x 5 K-chunks of v_wmma_f32_16x16x32_f16, C starts at 0 ----
        const v8f zero = {};
        v8f acc[4];
#pragma unroll
        for (int g = 0; g < 4; ++g) {
            acc[g] = __builtin_amdgcn_wmma_f32_16x16x32_f16(
                false, A[0], false, Bw[g][0], (short)0, zero, false, false);
#pragma unroll
            for (int k = 1; k < 5; ++k) {
                acc[g] = __builtin_amdgcn_wmma_f32_16x16x32_f16(
                    false, A[k], false, Bw[g][k], (short)0, acc[g], false, false);
            }
        }

        // ---- elementwise LSTM cell update (f32) ----
#pragma unroll
        for (int r = 0; r < 8; ++r) {
            const float fg = sigmoid_f(acc[0][r]);
            const float ig = sigmoid_f(acc[1][r]);
            const float ng = tanh_f(acc[2][r]);
            const float og = sigmoid_f(acc[3][r]);
            const float c  = cstate[r] * fg + ig * ng;
            cstate[r] = c;
            const float h = og * tanh_f(c);
            hout[r] = h;
            hreg[r] = (_Float16)h;
        }
        __syncthreads();   // all A reads done before next step overwrites gbuf
    }

    // ---- write final hidden state h_T [B, 128] f32 ----
#pragma unroll
    for (int r = 0; r < 8; ++r) {
        const int m = r + half * 8;
        out[(row0 + m) * (long)H_DIM + colbase + l16] = hout[r];
    }
}

extern "C" void kernel_launch(void* const* d_in, const int* in_sizes, int n_in,
                              void* d_out, int out_size, void* d_ws, size_t ws_size,
                              hipStream_t stream) {
    const float* x  = (const float*)d_in[0];
    const float* Wf = (const float*)d_in[1];
    const float* bf = (const float*)d_in[2];
    const float* Wi = (const float*)d_in[3];
    const float* bi = (const float*)d_in[4];
    const float* Wc = (const float*)d_in[5];
    const float* bc = (const float*)d_in[6];
    const float* Wo = (const float*)d_in[7];
    const float* bo = (const float*)d_in[8];
    float* out = (float*)d_out;

    const int nrows  = in_sizes[0] / (T_STEPS * IN_DIM);   // 65536
    const int blocks = nrows / MROWS;                       // 4096

    lstm_wmma_kernel<<<blocks, TPB, 0, stream>>>(
        x, Wf, bf, Wi, bi, Wc, bc, Wo, bo, out);
}
